// SentenceEmbeddingSimple_88484916232547
// MI455X (gfx1250) — compile-verified, matrix operand order checked by reading, not run on hardware
//
#include <hip/hip_runtime.h>

typedef __attribute__((ext_vector_type(16))) __bf16  v16bf;
typedef __attribute__((ext_vector_type(8)))  float   v8f;

#define VOCAB 2080
#define E     128
#define H     256
#define BSZ   4096
#define T     64
#define G     1024      // 4*H gate columns
#define KTOT  384       // E + H combined K
#define BT    16        // batch rows per block
#define AROW  400       // padded A-tile row (bf16 elems); 800 B keeps 32B alignment

__device__ __forceinline__ unsigned short f2bf(float f) {
  unsigned int u = __float_as_uint(f);
  u += 0x7FFFu + ((u >> 16) & 1u);          // round-to-nearest-even
  return (unsigned short)(u >> 16);
}
__device__ __forceinline__ float sigmoidf(float x) {
  return 1.0f / (1.0f + __expf(-x));
}
__device__ __forceinline__ float fast_tanhf(float x) {
  x = fminf(fmaxf(x, -15.0f), 15.0f);
  float e = __expf(2.0f * x);
  return (e - 1.0f) / (e + 1.0f);
}

// physical row p (0..1023) -> logical gate row in original [i;f;g;o] stacking.
// wave w = p>>7 owns 128 consecutive rows ordered [i0 i1 f0 f1 g0 g1 o0 o1],
// covering h-columns j in [32w, 32w+32).
__device__ __forceinline__ int perm_logical_row(int p) {
  int w    = p >> 7;
  int rem  = p & 127;
  int tile = rem >> 4;      // 0..7
  int r    = rem & 15;
  int gate = tile >> 1;     // 0..3 = i,f,g,o
  int s    = tile & 1;      // 16-col slot
  int j    = w * 32 + s * 16 + r;
  return gate * 256 + j;
}

// ---------------- prep kernels ----------------
__global__ void conv_emb_bf16(const float* __restrict__ src,
                              unsigned short* __restrict__ dst, int n) {
  int i = blockIdx.x * blockDim.x + threadIdx.x;
  if (i < n) dst[i] = f2bf(src[i]);
}

__global__ void build_wperm_bf16(const float* __restrict__ wih,
                                 const float* __restrict__ whh,
                                 unsigned short* __restrict__ dst) {
  int i = blockIdx.x * blockDim.x + threadIdx.x;
  if (i >= G * KTOT) return;
  int p = i / KTOT, k = i - p * KTOT;
  int L = perm_logical_row(p);
  float v = (k < E) ? wih[L * E + k] : whh[L * H + (k - E)];
  dst[i] = f2bf(v);
}

__global__ void build_bias_perm(const float* __restrict__ bih,
                                const float* __restrict__ bhh,
                                float* __restrict__ dst) {
  int p = blockIdx.x * blockDim.x + threadIdx.x;
  if (p < G) {
    int L = perm_logical_row(p);
    dst[p] = bih[L] + bhh[L];
  }
}

// ---------------- fused persistent LSTM ----------------
__global__ __launch_bounds__(256) void lstm_fused_kernel(
    const int* __restrict__ word_ids,          // (B, T)
    const int* __restrict__ lengths,           // (B,)
    const unsigned short* __restrict__ emb16,  // (VOCAB, E) bf16
    const unsigned short* __restrict__ wperm,  // (G, KTOT) bf16, gate-interleaved rows
    const float* __restrict__ biasp,           // (G,) permuted b_ih+b_hh
    float* __restrict__ out)                   // (B, H)
{
  __shared__ unsigned short a_tile[BT * AROW];   // [x (0..127) | h (128..383)] bf16

  const int tid  = threadIdx.x;
  const int lane = tid & 31;
  const int wave = tid >> 5;
  const int b0   = blockIdx.x * BT;
  const int lr   = lane & 15;      // row/col id within 16
  const int lh   = lane >> 4;      // lane half

  // gather mapping: 16 threads per batch row, 16B segments
  const int grow = tid >> 4;
  const int gseg = tid & 15;

  // hoisted per-lane constants
  float bias_reg[8];
  #pragma unroll
  for (int nt = 0; nt < 8; ++nt)
    bias_reg[nt] = biasp[wave * 128 + nt * 16 + lr];

  int len8[8];
  #pragma unroll
  for (int r = 0; r < 8; ++r)
    len8[r] = lengths[b0 + r + 8 * lh];

  float c_reg[2][8], h_reg[2][8];
  #pragma unroll
  for (int s = 0; s < 2; ++s)
    #pragma unroll
    for (int r = 0; r < 8; ++r) { c_reg[s][r] = 0.0f; h_reg[s][r] = 0.0f; }

  // zero h region of A tile (cols E..KTOT)
  #pragma unroll
  for (int k = 0; k < 16; ++k)
    a_tile[grow * AROW + E + gseg + k * 16] = 0;

  // x(0)
  uint4 xv;
  {
    int wid = word_ids[(b0 + grow) * T + 0];
    xv = ((const uint4*)(emb16 + (long)wid * E))[gseg];
  }
  *(uint4*)&a_tile[grow * AROW + gseg * 8] = xv;
  __syncthreads();

  v8f acc[8];
  int topa = 0;   // opaque zero offset, redefined every timestep

  for (int t = 0; t < T; ++t) {
    // Opaque SGPR offset: blocks loop-invariant hoisting (-> register spill
    // disaster) while keeping the pointer chain rooted at the kernel arg so
    // address-space inference still emits global_load (not flat_load).
    asm volatile("" : "+s"(topa));
    const unsigned short* wb = wperm + topa;

    #pragma unroll
    for (int nt = 0; nt < 8; ++nt)
      #pragma unroll
      for (int r = 0; r < 8; ++r) acc[nt][r] = 0.0f;

    // gates(phys) = [x|h] @ Wperm^T : 12 K-chunks x 8 N-tiles per wave
    #pragma unroll
    for (int kc = 0; kc < KTOT / 32; ++kc) {
      v16bf a = *(const v16bf*)&a_tile[lr * AROW + kc * 32 + lh * 16];
      #pragma unroll
      for (int nt = 0; nt < 8; ++nt) {
        v16bf b = *(const v16bf*)(wb + (long)(wave * 128 + nt * 16 + lr) * KTOT
                                     + kc * 32 + lh * 16);
        acc[nt] = __builtin_amdgcn_wmma_f32_16x16x32_bf16(
            false, a, false, b, (short)0, acc[nt], false, false);
      }
    }

    // prefetch next step's embedding segment (overlaps with WMMA latency)
    {
      int tt  = (t + 1 < T) ? (t + 1) : (T - 1);
      int wid = word_ids[(b0 + grow) * T + tt];
      xv = ((const uint4*)(emb16 + (long)wid * E))[gseg];
    }

    // activation straight out of accumulator registers:
    // acc[0..1]=i, acc[2..3]=f, acc[4..5]=g, acc[6..7]=o
    unsigned short hb[2][8];
    #pragma unroll
    for (int s = 0; s < 2; ++s)
      #pragma unroll
      for (int r = 0; r < 8; ++r) {
        float ig = acc[0 + s][r] + bias_reg[0 + s];
        float fg = acc[2 + s][r] + bias_reg[2 + s];
        float gg = acc[4 + s][r] + bias_reg[4 + s];
        float og = acc[6 + s][r] + bias_reg[6 + s];
        float cn = sigmoidf(fg) * c_reg[s][r] + sigmoidf(ig) * fast_tanhf(gg);
        float hn = sigmoidf(og) * fast_tanhf(cn);
        bool msk = (t < len8[r]);
        if (msk) { c_reg[s][r] = cn; h_reg[s][r] = hn; }
        hb[s][r] = f2bf(h_reg[s][r]);
      }

    __syncthreads();   // everyone is done reading a_tile

    // write x(t+1) and h(t) for the next step
    *(uint4*)&a_tile[grow * AROW + gseg * 8] = xv;
    #pragma unroll
    for (int s = 0; s < 2; ++s)
      #pragma unroll
      for (int r = 0; r < 8; ++r) {
        int M = r + 8 * lh;
        int j = wave * 32 + s * 16 + lr;
        a_tile[M * AROW + E + j] = hb[s][r];
      }
    __syncthreads();   // visible for next iteration
  }

  // final h_T
  #pragma unroll
  for (int s = 0; s < 2; ++s)
    #pragma unroll
    for (int r = 0; r < 8; ++r) {
      int M = r + 8 * lh;
      int j = wave * 32 + s * 16 + lr;
      out[(long)(b0 + M) * H + j] = h_reg[s][r];
    }
}

// ---------------- launcher ----------------
extern "C" void kernel_launch(void* const* d_in, const int* in_sizes, int n_in,
                              void* d_out, int out_size, void* d_ws, size_t ws_size,
                              hipStream_t stream) {
  const int*   word_ids = (const int*)  d_in[0];   // (B,T) int32
  const int*   lengths  = (const int*)  d_in[1];   // (B,)  int32
  const float* emb      = (const float*)d_in[2];   // (VOCAB,E)
  const float* W_ih     = (const float*)d_in[3];   // (4H,E)
  const float* W_hh     = (const float*)d_in[4];   // (4H,H)
  const float* b_ih     = (const float*)d_in[5];   // (4H,)
  const float* b_hh     = (const float*)d_in[6];   // (4H,)
  float* out = (float*)d_out;                      // (B,H)

  // workspace layout (all 32B aligned)
  unsigned char* ws = (unsigned char*)d_ws;
  unsigned short* emb16 = (unsigned short*)(ws);                  // 532480 B
  unsigned short* wperm = (unsigned short*)(ws + 532480);         // 786432 B
  float*          biasp = (float*)(ws + 532480 + 786432);         // 4096 B

  {
    int n = VOCAB * E;
    conv_emb_bf16<<<(n + 255) / 256, 256, 0, stream>>>(emb, emb16, n);
  }
  {
    int n = G * KTOT;
    build_wperm_bf16<<<(n + 255) / 256, 256, 0, stream>>>(W_ih, W_hh, wperm);
  }
  build_bias_perm<<<(G + 255) / 256, 256, 0, stream>>>(b_ih, b_hh, biasp);

  lstm_fused_kernel<<<BSZ / BT, 256, 0, stream>>>(word_ids, lengths, emb16,
                                                  wperm, biasp, out);
}
